// GlobalFractalRouter_76218489634881
// MI455X (gfx1250) — matrix operand order repaired and checked by reading, not compiled, not run on hardware
//
#include <hip/hip_runtime.h>
#include <math.h>

#define NTOK 8192
#define DDIM 1024
#define NEXP 64
#define FPDIM 128
#define HDIM 256
#define NFDIM 4
#define TOPK 2
#define MROWS (NTOK * TOPK)   // 16384

typedef __attribute__((ext_vector_type(16))) __bf16 v16bf;
typedef __attribute__((ext_vector_type(8)))  float  v8f;

__device__ __forceinline__ __bf16 f2bf(float f) {
  unsigned u = __builtin_bit_cast(unsigned, f);
  unsigned r = u + 0x7FFFu + ((u >> 16) & 1u);   // round-to-nearest-even
  unsigned short h = (unsigned short)(r >> 16);
  return __builtin_bit_cast(__bf16, h);
}
__device__ __forceinline__ float geluf(float x) {
  return 0.5f * x * (1.0f + erff(x * 0.70710678118654752f));
}
__device__ __forceinline__ float sigmf(float x) { return 1.0f / (1.0f + expf(-x)); }

#define WMMA_BF16(a, b, c) \
  __builtin_amdgcn_wmma_f32_16x16x32_bf16(false, (a), false, (b), (short)0, (c), false, false)

// ---------------------------------------------------------------------------
// A-fragment: row-major bf16 [.,lda]; lane holds M=lane%16,
//  K chunks [ks+(lane/16)*8 .. +7] and [ks+16+(lane/16)*8 .. +7]
__device__ __forceinline__ v16bf frag_a(const __bf16* __restrict__ A, int lda,
                                        int row, int koff, int ks) {
  const __bf16* p = A + (size_t)row * lda + ks + koff;
  v16bf a;
  ((float4*)&a)[0] = *(const float4*)p;
  ((float4*)&a)[1] = *(const float4*)(p + 16);
  return a;
}
// B-fragment: B stored transposed [N][K] bf16; lane holds N=lane%16,
//  K chunks [ks+(lane/16)*8 .. +7] and [ks+16+(lane/16)*8 .. +7]
__device__ __forceinline__ v16bf frag_b(const __bf16* __restrict__ Bt, int ldb,
                                        int col, int khalf, int ks) {
  const __bf16* p = Bt + (size_t)col * ldb + ks + khalf / 2;
  v16bf b;
  ((float4*)&b)[0] = *(const float4*)p;
  ((float4*)&b)[1] = *(const float4*)(p + 16);
  return b;
}

// ---------------------------------------------------------------------------
// 1) fingerprints fp[E][FPD], L2-normalized rows
__global__ void k_fp(float* __restrict__ fp) {
  __shared__ float red[FPDIM];
  const int i = blockIdx.x, d = threadIdx.x;
  const double theta = 6.283185307179586 * (double)i / 1.6180339887498949;
  const float v = (float)sin(theta * (double)(d + 1) / (double)FPDIM);
  red[d] = v * v;
  __syncthreads();
  for (int s = FPDIM / 2; s > 0; s >>= 1) {
    if (d < s) red[d] += red[d + s];
    __syncthreads();
  }
  fp[i * FPDIM + d] = v / sqrtf(red[0]);
}

// 2) keys_bf16[E][D] = emb@Wq + bq + fp@Wfp + bfp  (B-transposed layout for WMMA)
__global__ void k_keys(const float* __restrict__ emb, const float* __restrict__ Wq,
                       const float* __restrict__ bq, const float* __restrict__ fp,
                       const float* __restrict__ Wfp, const float* __restrict__ bfp,
                       __bf16* __restrict__ keysb) {
  const int e = blockIdx.x;
  const int d = blockIdx.y * blockDim.x + threadIdx.x;
  float acc = bq[d] + bfp[d];
  const float* er = emb + e * DDIM;
  for (int k = 0; k < DDIM; ++k) acc += er[k] * Wq[k * DDIM + d];
  const float* fr = fp + e * FPDIM;
  for (int j = 0; j < FPDIM; ++j) acc += fr[j] * Wfp[j * DDIM + d];
  keysb[e * DDIM + d] = f2bf(acc);
}

// 3) fp_h[E][H], compat[E][NF], src_h[H]=src_fp@W1f, src_gc[H]=g1b+src_fp@G1[8:136]
__global__ void k_small(const float* __restrict__ fp, const float* __restrict__ W1f,
                        const float* __restrict__ src_fp, const float* __restrict__ Wc,
                        const float* __restrict__ bc, const float* __restrict__ G1,
                        const float* __restrict__ g1b, float* __restrict__ fph,
                        float* __restrict__ compat, float* __restrict__ srch,
                        float* __restrict__ srcgc) {
  const int e = blockIdx.x, h = threadIdx.x;
  float acc = 0.f;
  const float* fr = fp + e * FPDIM;
  for (int j = 0; j < FPDIM; ++j) acc += fr[j] * W1f[j * HDIM + h];
  fph[e * HDIM + h] = acc;
  if (h < NFDIM) {
    float c = bc[h];
    for (int j = 0; j < FPDIM; ++j) c += src_fp[j] * Wc[j * NFDIM + h];
    for (int j = 0; j < FPDIM; ++j) c += fr[j] * Wc[(FPDIM + j) * NFDIM + h];
    compat[e * NFDIM + h] = tanhf(c);
  }
  if (e == 0) {
    float s = 0.f, g = g1b[h];
    for (int j = 0; j < FPDIM; ++j) {
      s += src_fp[j] * W1f[j * HDIM + h];
      g += src_fp[j] * G1[(2 * NFDIM + j) * HDIM + h];
    }
    srch[h] = s;
    srcgc[h] = g;
  }
}

// 4) f32 -> bf16 bulk convert
__global__ void k_cvt_bf16(const float* __restrict__ src, __bf16* __restrict__ dst,
                           int ntot) {
  const int i = blockIdx.x * blockDim.x + threadIdx.x;
  if (i < ntot) dst[i] = f2bf(src[i]);
}

// 5) transpose f32 [R][C] -> bf16 [C][R]
__global__ void k_transpose_bf16(const float* __restrict__ src, __bf16* __restrict__ dst,
                                 int R, int C) {
  const int i = blockIdx.x * blockDim.x + threadIdx.x;
  if (i < R * C) {
    const int r = i / C, c = i % C;
    dst[c * R + r] = f2bf(src[i]);
  }
}

// 6) logits = x @ keys^T : M=8192 N=64 K=1024
//    block = 64 (2 waves); wave handles 1 M-tile x 2 E-tiles (A reused twice)
__global__ void k_gemm_logits(const __bf16* __restrict__ xb,
                              const __bf16* __restrict__ keysb,
                              float* __restrict__ logits) {
  const int lane = threadIdx.x & 31;
  const int wave = threadIdx.x >> 5;        // 0..1 -> e tiles {2w, 2w+1}
  const int m_base = blockIdx.x * 16;
  const int e0 = wave * 32;
  const int arow = m_base + (lane & 15);
  const int akoff = (lane >> 4) * 8;
  const int bcol0 = e0 + (lane & 15);
  const int bkoff = (lane >> 4) * 16;
  v8f acc0 = {}, acc1 = {};
  for (int ks = 0; ks < DDIM; ks += 32) {
    __builtin_prefetch(xb + (size_t)arow * DDIM + ks + 64, 0, 3);
    v16bf a  = frag_a(xb, DDIM, arow, akoff, ks);
    v16bf b0 = frag_b(keysb, DDIM, bcol0, bkoff, ks);
    v16bf b1 = frag_b(keysb, DDIM, bcol0 + 16, bkoff, ks);
    acc0 = WMMA_BF16(a, b0, acc0);
    acc1 = WMMA_BF16(a, b1, acc1);
  }
  const int hi = lane >> 4;
  for (int r = 0; r < 8; ++r) {
    const size_t row = (size_t)(m_base + 8 * hi + r) * NEXP;
    logits[row + e0 + (lane & 15)]      = acc0[r];
    logits[row + e0 + 16 + (lane & 15)] = acc1[r];
  }
}

// 7) top-2 + softmax weights
__global__ void k_topk(const float* __restrict__ logits, int* __restrict__ tidx,
                       float* __restrict__ tw) {
  const int n = blockIdx.x * blockDim.x + threadIdx.x;
  const float* L = logits + (size_t)n * NEXP;
  float v0 = -INFINITY, v1 = -INFINITY;
  int i0 = 0, i1 = 0;
  for (int e = 0; e < NEXP; ++e) {
    const float v = L[e];
    if (v > v0) { v1 = v0; i1 = i0; v0 = v; i0 = e; }
    else if (v > v1) { v1 = v; i1 = e; }
  }
  const float eo = expf(v1 - v0);
  const float inv = 1.f / (1.f + eo);
  tidx[n * TOPK + 0] = i0;
  tidx[n * TOPK + 1] = i1;
  tw[n * TOPK + 0] = inv;
  tw[n * TOPK + 1] = eo * inv;
}

// 8) xh = x @ W1x : M=8192 N=256 K=1024
//    block = 256 (8 waves); wave = 1 M-tile x 2 N-tiles (A fragment reused 2x)
__global__ void k_gemm_xh(const __bf16* __restrict__ xb, const __bf16* __restrict__ w1xt,
                          float* __restrict__ xh) {
  const int lane = threadIdx.x & 31;
  const int wave = threadIdx.x >> 5;        // 0..7
  const int m_base = blockIdx.x * 32 + (wave & 1) * 16;
  const int h_base = blockIdx.y * 128 + (wave >> 1) * 32;
  const int arow0 = m_base + (lane & 15);
  const int akoff = (lane >> 4) * 8;
  const int bcol0 = h_base + (lane & 15);
  const int bkoff = (lane >> 4) * 16;
  v8f acc00 = {}, acc01 = {};
  for (int ks = 0; ks < DDIM; ks += 32) {
    v16bf a0 = frag_a(xb, DDIM, arow0, akoff, ks);
    v16bf b0 = frag_b(w1xt, DDIM, bcol0, bkoff, ks);
    v16bf b1 = frag_b(w1xt, DDIM, bcol0 + 16, bkoff, ks);
    acc00 = WMMA_BF16(a0, b0, acc00);
    acc01 = WMMA_BF16(a0, b1, acc01);
  }
  const int hi = lane >> 4;
  for (int r = 0; r < 8; ++r) {
    const size_t row = (size_t)(m_base + 8 * hi + r) * HDIM;
    xh[row + bcol0]      = acc00[r];
    xh[row + bcol0 + 16] = acc01[r];
  }
}

// 9) hidden[m][H] = gelu(grad@G1[0:4] + compat@G1[4:8] + src_gc)  (bf16)
//    one wave per row m; lanes own h = lane + 32j (coalesced); pots reduced
//    with shfl_xor butterflies.
__global__ void k_hidden(const float* __restrict__ xh, const float* __restrict__ fph,
                         const float* __restrict__ srch, const float* __restrict__ srcgc,
                         const float* __restrict__ b1, const float* __restrict__ W2,
                         const float* __restrict__ compat, const float* __restrict__ G1,
                         const int* __restrict__ tidx, __bf16* __restrict__ hidden) {
  const int wave = threadIdx.x >> 5, lane = threadIdx.x & 31;
  const int m = blockIdx.x * 8 + wave;
  const int n = m >> 1, k = m & 1;
  const int e = tidx[n * TOPK + k];
  const float* xr = xh + (size_t)n * HDIM;
  const float* fr = fph + (size_t)e * HDIM;
  float sp[NFDIM] = {0, 0, 0, 0}, tp[NFDIM] = {0, 0, 0, 0};
  float sarr[8], tarr[8];
  for (int j = 0; j < 8; ++j) {
    const int h = lane + 32 * j;
    const float xv = xr[h], bv = b1[h];
    const float s = geluf(xv + srch[h] + bv);
    const float t = geluf(xv + fr[h] + bv);
    sarr[j] = s; tarr[j] = t;
    const float* w2 = W2 + h * NFDIM;
    for (int f = 0; f < NFDIM; ++f) { sp[f] += s * w2[f]; tp[f] += t * w2[f]; }
  }
  (void)sarr; (void)tarr;
  for (int off = 16; off > 0; off >>= 1)
    for (int f = 0; f < NFDIM; ++f) {
      sp[f] += __shfl_xor(sp[f], off, 32);
      tp[f] += __shfl_xor(tp[f], off, 32);
    }
  float gr[NFDIM], cc[NFDIM];
  for (int f = 0; f < NFDIM; ++f) {
    gr[f] = tp[f] - sp[f];                 // b2 cancels in tgt_pot - src_pot
    cc[f] = compat[e * NFDIM + f];
  }
  __bf16* hr = hidden + (size_t)m * HDIM;
  for (int j = 0; j < 8; ++j) {
    const int h = lane + 32 * j;
    float pre = srcgc[h];
    for (int f = 0; f < NFDIM; ++f)
      pre += gr[f] * G1[f * HDIM + h] + cc[f] * G1[(NFDIM + f) * HDIM + h];
    hr[h] = f2bf(geluf(pre));
  }
}

// 10) out[n][d] = sum_k w[n,k] * x[n,d] * sigmoid(hidden[2n+k]@G2[:,d] + g2b[d])
//     M=16384 N=1024 K=256; block=256 (8 waves); wave = 1 M-tile x 2 D-tiles.
//     Rows 2n,2n+1 of a tile live in adjacent accumulator VGPRs of the same lane,
//     so the top-k combine is a free in-register add (no atomics).
__global__ void k_gemm_gate(const __bf16* __restrict__ hidden,
                            const __bf16* __restrict__ g2t,
                            const float* __restrict__ xs, const float* __restrict__ tw,
                            const float* __restrict__ g2b, float* __restrict__ out) {
  const int lane = threadIdx.x & 31;
  const int wave = threadIdx.x >> 5;        // 0..7
  const int m_base = blockIdx.x * 32 + (wave & 1) * 16;
  const int d_base = blockIdx.y * 128 + (wave >> 1) * 32;
  const int arow0 = m_base + (lane & 15);
  const int akoff = (lane >> 4) * 8;
  const int bcol0 = d_base + (lane & 15);
  const int bkoff = (lane >> 4) * 16;
  v8f acc0 = {}, acc1 = {};
  for (int ks = 0; ks < HDIM; ks += 32) {
    __builtin_prefetch(hidden + (size_t)arow0 * HDIM + ks + 32, 0, 3);
    v16bf a  = frag_a(hidden, HDIM, arow0, akoff, ks);
    v16bf b0 = frag_b(g2t, HDIM, bcol0, bkoff, ks);
    v16bf b1 = frag_b(g2t, HDIM, bcol0 + 16, bkoff, ks);
    acc0 = WMMA_BF16(a, b0, acc0);
    acc1 = WMMA_BF16(a, b1, acc1);
  }
  const int hi = lane >> 4;
  for (int t = 0; t < 2; ++t) {
    const v8f acc = t ? acc1 : acc0;
    const int col = bcol0 + t * 16;
    const float gb = g2b[col];
    for (int a = 0; a < 4; ++a) {
      const int m0 = m_base + 8 * hi + 2 * a;   // k=0 row (m_base even -> parity == k)
      const int n = m0 >> 1;
      const float g0 = sigmf(acc[2 * a] + gb);
      const float g1 = sigmf(acc[2 * a + 1] + gb);
      const float xv = xs[(size_t)n * DDIM + col];
      out[(size_t)n * DDIM + col] = xv * (tw[n * TOPK] * g0 + tw[n * TOPK + 1] * g1);
    }
  }
}

// ---------------------------------------------------------------------------
extern "C" void kernel_launch(void* const* d_in, const int* in_sizes, int n_in,
                              void* d_out, int out_size, void* d_ws, size_t ws_size,
                              hipStream_t stream) {
  (void)in_sizes; (void)n_in; (void)out_size; (void)ws_size;
  const float* x      = (const float*)d_in[0];
  const float* emb    = (const float*)d_in[1];
  const float* Wq     = (const float*)d_in[2];
  const float* bq     = (const float*)d_in[3];
  const float* Wfp    = (const float*)d_in[4];
  const float* bfp    = (const float*)d_in[5];
  const float* src_fp = (const float*)d_in[6];
  const float* W1x    = (const float*)d_in[7];
  const float* W1f    = (const float*)d_in[8];
  const float* b1     = (const float*)d_in[9];
  const float* W2     = (const float*)d_in[10];
  // d_in[11] = b2 : cancels in gradient = tgt_pot - src_pot
  const float* Wc     = (const float*)d_in[12];
  const float* bc     = (const float*)d_in[13];
  const float* G1     = (const float*)d_in[14];
  const float* g1b    = (const float*)d_in[15];
  const float* G2     = (const float*)d_in[16];
  const float* g2b    = (const float*)d_in[17];
  // d_in[18] = top_k == 2 (compile-time TOPK)
  float* out = (float*)d_out;

  char* ws = (char*)d_ws;
  size_t off = 0;
  auto alloc = [&](size_t bytes) { void* p = ws + off; off = (off + bytes + 255) & ~(size_t)255; return p; };
  float*  fp_buf = (float*) alloc((size_t)NEXP * FPDIM * 4);
  __bf16* keysb  = (__bf16*)alloc((size_t)NEXP * DDIM * 2);
  float*  fph    = (float*) alloc((size_t)NEXP * HDIM * 4);
  float*  compat = (float*) alloc((size_t)NEXP * NFDIM * 4);
  float*  srch   = (float*) alloc((size_t)HDIM * 4);
  float*  srcgc  = (float*) alloc((size_t)HDIM * 4);
  __bf16* xb     = (__bf16*)alloc((size_t)NTOK * DDIM * 2);
  __bf16* w1xt   = (__bf16*)alloc((size_t)HDIM * DDIM * 2);
  __bf16* g2t    = (__bf16*)alloc((size_t)DDIM * HDIM * 2);
  float*  logits = (float*) alloc((size_t)NTOK * NEXP * 4);
  int*    tidx   = (int*)   alloc((size_t)NTOK * TOPK * 4);
  float*  tw     = (float*) alloc((size_t)NTOK * TOPK * 4);
  float*  xh     = (float*) alloc((size_t)NTOK * HDIM * 4);
  __bf16* hidden = (__bf16*)alloc((size_t)MROWS * HDIM * 2);

  // setup (tiny)
  k_fp<<<NEXP, FPDIM, 0, stream>>>(fp_buf);
  k_keys<<<dim3(NEXP, DDIM / 256), 256, 0, stream>>>(emb, Wq, bq, fp_buf, Wfp, bfp, keysb);
  k_small<<<NEXP, HDIM, 0, stream>>>(fp_buf, W1f, src_fp, Wc, bc, G1, g1b,
                                     fph, compat, srch, srcgc);
  k_cvt_bf16<<<(NTOK * DDIM) / 256, 256, 0, stream>>>(x, xb, NTOK * DDIM);
  k_transpose_bf16<<<(DDIM * HDIM) / 256, 256, 0, stream>>>(W1x, w1xt, DDIM, HDIM);
  k_transpose_bf16<<<(HDIM * DDIM) / 256, 256, 0, stream>>>(G2, g2t, HDIM, DDIM);

  // routing
  k_gemm_logits<<<NTOK / 16, 64, 0, stream>>>(xb, keysb, logits);
  k_topk<<<NTOK / 256, 256, 0, stream>>>(logits, tidx, tw);

  // field net
  k_gemm_xh<<<dim3(NTOK / 32, HDIM / 128), 256, 0, stream>>>(xb, w1xt, xh);
  k_hidden<<<MROWS / 8, 256, 0, stream>>>(xh, fph, srch, srcgc, b1, W2, compat,
                                          G1, tidx, hidden);

  // gate GEMM + fused sigmoid + dispatch-weighted combine
  k_gemm_gate<<<dim3(MROWS / 32, DDIM / 128), 256, 0, stream>>>(hidden, g2t, x, tw,
                                                                g2b, out);
}
